// EncoderSISO_67465346286286
// MI455X (gfx1250) — matrix operand order repaired and checked by reading, not compiled, not run on hardware
//
#include <hip/hip_runtime.h>
#include <hip/hip_bf16.h>
#include <math.h>

typedef __attribute__((ext_vector_type(16))) _Float16 v16h;
typedef __attribute__((ext_vector_type(8)))  float    v8f;

#define EPS_BN 1e-3f
#define BSZ  256
#define LSZ  4096
#define DIM_ 256
#define L3_  (3*LSZ)   // 12288

// ---- workspace layout (float offsets) ----
#define WS_YT    ((size_t)0)                        // [BSZ*L3_] interleaved e1,e2,eI
#define WS_PART1 (WS_YT + (size_t)BSZ*L3_)          // [256][1536] per-block channel stats
#define WS_PART2 (WS_PART1 + (size_t)256*1536)      // [4096][2]  yt stat partials
#define WS_PART3 (WS_PART2 + (size_t)4096*2)        // [12288][4] head stat partials
#define WS_FOLDA (WS_PART3 + (size_t)12288*4)       // [3][5][256] folded conv weights
#define WS_FOLDC (WS_FOLDA + (size_t)3*5*256)       // [3][256]    folded conv consts
#define WS_MID   (WS_FOLDC + (size_t)3*256)         // alpha[256], beta[256]
#define WS_OUTF  (WS_MID + (size_t)512)             // sc0, sc1, sh0, sh1

__device__ __forceinline__ float elu_f(float a) {
  return a > 0.f ? a : (__expf(a) - 1.f);
}

// ---------------------------------------------------------------------------
// Kernel 1: per-channel sum / sumsq of conv outputs for units 1,2,I.
// grid = BSZ blocks (one batch row), block = 256 threads (one channel each).
// ---------------------------------------------------------------------------
__global__ __launch_bounds__(256)
void k_stats(const float* __restrict__ x, const int* __restrict__ perms,
             const float* __restrict__ Wc1, const float* __restrict__ bc1,
             const float* __restrict__ Wc2, const float* __restrict__ bc2,
             const float* __restrict__ WcI, const float* __restrict__ bcI,
             float* __restrict__ ws)
{
  __shared__ float xs[260];
  __shared__ float xI[260];
  const int b = blockIdx.x;
  const int c = threadIdx.x;
  const float* xrow = x + (size_t)b * LSZ;
  const int*   prow = perms + (size_t)b * LSZ;

  float w1[5], w2[5], wI[5];
#pragma unroll
  for (int k = 0; k < 5; ++k) {
    w1[k] = Wc1[k*DIM_ + c];
    w2[k] = Wc2[k*DIM_ + c];
    wI[k] = WcI[k*DIM_ + c];
  }
  const float B1 = bc1[c], B2 = bc2[c], BI = bcI[c];
  float s1 = 0.f, q1 = 0.f, s2 = 0.f, q2 = 0.f, sI = 0.f, qI = 0.f;

  for (int tile = 0; tile < LSZ; tile += 256) {
    __syncthreads();
    for (int i = c; i < 260; i += 256) {
      const int gt = tile + i - 2;
      float v = 0.f, vi = 0.f;
      if (gt >= 0 && gt < LSZ) {
        v  = 2.f * xrow[gt] - 1.f;
        vi = 2.f * xrow[prow[gt]] - 1.f;
      }
      xs[i] = v; xI[i] = vi;
    }
    if (tile + 256 < LSZ) __builtin_prefetch(xrow + tile + 256, 0, 1);
    __syncthreads();
#pragma unroll 2
    for (int j = 0; j < 256; ++j) {
      const float a0 = xs[j], a1 = xs[j+1], a2 = xs[j+2], a3 = xs[j+3], a4 = xs[j+4];
      const float i0 = xI[j], i1 = xI[j+1], i2 = xI[j+2], i3 = xI[j+3], i4 = xI[j+4];
      float y1 = fmaf(w1[4], a4, fmaf(w1[3], a3, fmaf(w1[2], a2, fmaf(w1[1], a1, fmaf(w1[0], a0, B1)))));
      float y2 = fmaf(w2[4], a4, fmaf(w2[3], a3, fmaf(w2[2], a2, fmaf(w2[1], a1, fmaf(w2[0], a0, B2)))));
      float yI = fmaf(wI[4], i4, fmaf(wI[3], i3, fmaf(wI[2], i2, fmaf(wI[1], i1, fmaf(wI[0], i0, BI)))));
      s1 += y1; q1 = fmaf(y1, y1, q1);
      s2 += y2; q2 = fmaf(y2, y2, q2);
      sI += yI; qI = fmaf(yI, yI, qI);
    }
  }
  float* p = ws + WS_PART1 + (size_t)b * 1536;
  p[c]        = s1;  p[256 + c]  = s2;  p[512 + c]  = sI;
  p[768 + c]  = q1;  p[1024 + c] = q2;  p[1280 + c] = qI;
}

// ---------------------------------------------------------------------------
// Kernel 2: finalize BN stats (deterministic sum over 256 partials) and fold
// BN scale/shift into conv weights.  grid = 3 (unit), block = 256 (channel).
// ---------------------------------------------------------------------------
__global__ __launch_bounds__(256)
void k_fold(const float* Wc1, const float* bc1, const float* g1, const float* be1,
            const float* Wc2, const float* bc2, const float* g2, const float* be2,
            const float* WcI, const float* bcI, const float* gI, const float* beI,
            float* __restrict__ ws)
{
  const int u = blockIdx.x, c = threadIdx.x;
  const float* Wc = (u == 0) ? Wc1 : ((u == 1) ? Wc2 : WcI);
  const float* bc = (u == 0) ? bc1 : ((u == 1) ? bc2 : bcI);
  const float* g  = (u == 0) ? g1  : ((u == 1) ? g2  : gI);
  const float* be = (u == 0) ? be1 : ((u == 1) ? be2 : beI);

  float s = 0.f, q = 0.f;
  for (int p = 0; p < 256; ++p) {
    const float* pp = ws + WS_PART1 + (size_t)p * 1536;
    s += pp[u*256 + c];
    q += pp[768 + u*256 + c];
  }
  const float inv = 1.f / ((float)BSZ * (float)LSZ);
  const float m  = s * inv;
  const float v  = q * inv - m * m;
  const float sc = g[c] * rsqrtf(v + EPS_BN);
  const float sh = be[c] - sc * m;
#pragma unroll
  for (int k = 0; k < 5; ++k)
    ws[WS_FOLDA + (size_t)(u*5 + k)*DIM_ + c] = sc * Wc[k*DIM_ + c];
  ws[WS_FOLDC + (size_t)u*DIM_ + c] = sc * bc[c] + sh;
}

// ---------------------------------------------------------------------------
// Kernel 3: fused conv+BN+ELU+projection for all 3 units via WMMA.
// A = Wf in row 0 (16x32 f16), B = ELU activations (32 ch x 16 positions),
// 8 WMMAs accumulate K=256.  Writes interleaved yt and yt-stat partials.
// grid = (LSZ/256, BSZ), block = 256 (8 waves, 32 positions per wave).
// ---------------------------------------------------------------------------
__global__ __launch_bounds__(256)
void k_encode(const float* __restrict__ x, const int* __restrict__ perms,
              const float* __restrict__ Wf1, const float* __restrict__ bf1,
              const float* __restrict__ Wf2, const float* __restrict__ bf2,
              const float* __restrict__ WfI, const float* __restrict__ bfI,
              float* __restrict__ ws)
{
  __shared__ float xs[260];
  __shared__ float xI[260];
  __shared__ float fA[3*5*256];
  __shared__ float fC[3*256];
  __shared__ float wf[3*256];
  __shared__ float red[256];

  const int b   = blockIdx.y;
  const int P0  = blockIdx.x * 256;
  const int tid = threadIdx.x;
  const float* xrow = x + (size_t)b * LSZ;
  const int*   prow = perms + (size_t)b * LSZ;

  for (int i = tid; i < 3*5*256; i += 256) fA[i] = ws[WS_FOLDA + i];
  for (int i = tid; i < 3*256;   i += 256) fC[i] = ws[WS_FOLDC + i];
  wf[tid] = Wf1[tid]; wf[256 + tid] = Wf2[tid]; wf[512 + tid] = WfI[tid];
  for (int i = tid; i < 260; i += 256) {
    const int gt = P0 + i - 2;
    float v = 0.f, vi = 0.f;
    if (gt >= 0 && gt < LSZ) {
      v  = 2.f * xrow[gt] - 1.f;
      vi = 2.f * xrow[prow[gt]] - 1.f;
    }
    xs[i] = v; xI[i] = vi;
  }
  __syncthreads();

  const int lane = tid & 31;
  const int wid  = tid >> 5;
  const int n    = lane & 15;      // position-in-slab / A-matrix row
  const int half = lane >> 4;      // K-half selector
  const float rowsel = (n == 0) ? 1.f : 0.f;   // arithmetic row-0 mask (no branches)
  const float bf1v = bf1[0], bf2v = bf2[0], bfIv = bfI[0];

  float ysum = 0.f, ysq = 0.f;
  for (int sl = 0; sl < 2; ++sl) {
    const int j = wid*32 + sl*16 + n;   // local position (0..255)
    float xw[5], xwI[5];
#pragma unroll
    for (int k = 0; k < 5; ++k) { xw[k] = xs[j + k]; xwI[k] = xI[j + k]; }

    for (int u = 0; u < 3; ++u) {
      // hoist window select out of the FMA nest (uniform per wave)
      float w5[5];
#pragma unroll
      for (int k = 0; k < 5; ++k) w5[k] = (u == 2) ? xwI[k] : xw[k];

      v8f acc = {};
      for (int ki = 0; ki < 8; ++ki) {
        // B fragment: ELU(conv) activations, channel = 32*ki + half*16 + i
        v16h bfrag;
#pragma unroll
        for (int i = 0; i < 16; ++i) {
          const int ch = 32*ki + half*16 + i;
          float a = fC[u*256 + ch];
#pragma unroll
          for (int k = 0; k < 5; ++k)
            a = fmaf(fA[(u*5 + k)*256 + ch], w5[k], a);
          bfrag[i] = (_Float16)elu_f(a);
        }
        // A fragment: Wf chunk in row 0 only.  Unconditional LDS broadcast
        // load, masked arithmetically -> v_mul + v_cvt_pk, no EXEC branching.
        v16h afrag;
#pragma unroll
        for (int i = 0; i < 16; ++i) {
          const int K = half*16 + i;
          const float wv_ = wf[u*256 + 32*ki + K];
          afrag[i] = (_Float16)(rowsel * wv_);
        }
        acc = __builtin_amdgcn_wmma_f32_16x16x32_f16(
            false, afrag, false, bfrag, (short)0, acc, false, false);
      }
      const float bfu = (u == 0) ? bf1v : ((u == 1) ? bf2v : bfIv);
      const float e = acc[0] + bfu;     // D[0, n] lives in lanes 0..15, VGPR0
      if (half == 0) {
        ws[WS_YT + (size_t)b*L3_ + (size_t)(3*(P0 + j) + u)] = e;
        ysum += e; ysq = fmaf(e, e, ysq);
      }
    }
  }

  // deterministic block reduction of yt partial stats
  red[tid] = ysum; __syncthreads();
  for (int o = 128; o > 0; o >>= 1) { if (tid < o) red[tid] += red[tid + o]; __syncthreads(); }
  const int blin = blockIdx.y * gridDim.x + blockIdx.x;
  if (tid == 0) ws[WS_PART2 + 2*(size_t)blin] = red[0];
  __syncthreads();
  red[tid] = ysq; __syncthreads();
  for (int o = 128; o > 0; o >>= 1) { if (tid < o) red[tid] += red[tid + o]; __syncthreads(); }
  if (tid == 0) ws[WS_PART2 + 2*(size_t)blin + 1] = red[0];
}

// ---------------------------------------------------------------------------
// Kernel 4: yt scalar stats -> closed-form middle-unit BN fold (alpha, beta).
// ---------------------------------------------------------------------------
__global__ __launch_bounds__(256)
void k_mid(const float* Wm, const float* bm, const float* gm, const float* bem,
           float* __restrict__ ws)
{
  __shared__ float red[256];
  __shared__ float st[2];
  const int tid = threadIdx.x;
  float s = 0.f, q = 0.f;
  for (int p = tid; p < 4096; p += 256) {
    s += ws[WS_PART2 + 2*(size_t)p];
    q += ws[WS_PART2 + 2*(size_t)p + 1];
  }
  red[tid] = s; __syncthreads();
  for (int o = 128; o > 0; o >>= 1) { if (tid < o) red[tid] += red[tid + o]; __syncthreads(); }
  if (tid == 0) st[0] = red[0];
  __syncthreads();
  red[tid] = q; __syncthreads();
  for (int o = 128; o > 0; o >>= 1) { if (tid < o) red[tid] += red[tid + o]; __syncthreads(); }
  if (tid == 0) st[1] = red[0];
  __syncthreads();

  const float inv = 1.f / ((float)BSZ * (float)L3_);
  const float my = st[0] * inv;
  const float vy = st[1] * inv - my * my;
  const int c = tid;
  const float w    = Wm[c];
  const float var  = vy * w * w;
  const float sc   = gm[c] * rsqrtf(var + EPS_BN);
  ws[WS_MID + c]       = sc * w;                 // alpha
  ws[WS_MID + 256 + c] = bem[c] - sc * my * w;   // beta
}

// ---------------------------------------------------------------------------
// Kernel 5: head: h_j = sum_c Wo[c,j]*ELU(alpha*yt+beta) + bo[j] via WMMA
// (rows 0,1 of A = Wo columns).  Writes pre-BN output to d_out + partials.
// grid = (L3_/256, BSZ), block = 256.
// ---------------------------------------------------------------------------
__global__ __launch_bounds__(256)
void k_head(const float* __restrict__ Wo, const float* __restrict__ bo,
            float* __restrict__ out, float* __restrict__ ws)
{
  __shared__ float al[256];
  __shared__ float bt[256];
  __shared__ float wo[512];
  __shared__ float red[256];

  const int b   = blockIdx.y;
  const int P0  = blockIdx.x * 256;
  const int tid = threadIdx.x;
  al[tid] = ws[WS_MID + tid];
  bt[tid] = ws[WS_MID + 256 + tid];
  wo[2*tid] = Wo[2*tid]; wo[2*tid + 1] = Wo[2*tid + 1];
  __syncthreads();

  const int lane = tid & 31;
  const int wid  = tid >> 5;
  const int n    = lane & 15;
  const int half = lane >> 4;
  const int ncl  = (n < 2) ? n : 0;            // clamped column for safe load
  const float rowsel = (n < 2) ? 1.f : 0.f;    // arithmetic rows-0/1 mask
  const float bo0 = bo[0], bo1 = bo[1];

  float s0 = 0.f, s1 = 0.f, q0 = 0.f, q1 = 0.f;
  for (int sl = 0; sl < 2; ++sl) {
    const int j = wid*32 + sl*16 + n;
    const float v = ws[WS_YT + (size_t)b*L3_ + (size_t)(P0 + j)];
    v8f acc = {};
    for (int ki = 0; ki < 8; ++ki) {
      v16h bfrag;
#pragma unroll
      for (int i = 0; i < 16; ++i) {
        const int ch = 32*ki + half*16 + i;
        bfrag[i] = (_Float16)elu_f(fmaf(al[ch], v, bt[ch]));
      }
      v16h afrag;
#pragma unroll
      for (int i = 0; i < 16; ++i) {
        const int K = half*16 + i;
        const float wv_ = wo[(32*ki + K)*2 + ncl];   // unconditional load
        afrag[i] = (_Float16)(rowsel * wv_);
      }
      acc = __builtin_amdgcn_wmma_f32_16x16x32_f16(
          false, afrag, false, bfrag, (short)0, acc, false, false);
    }
    if (half == 0) {
      const float h0 = acc[0] + bo0;   // D[0,n]
      const float h1 = acc[1] + bo1;   // D[1,n]
      const size_t o = ((size_t)b*L3_ + (size_t)(P0 + j)) * 2;
      out[o] = h0; out[o + 1] = h1;
      s0 += h0; q0 = fmaf(h0, h0, q0);
      s1 += h1; q1 = fmaf(h1, h1, q1);
    }
  }

  const int blin = blockIdx.y * gridDim.x + blockIdx.x;
  float vals[4] = { s0, s1, q0, q1 };
#pragma unroll
  for (int slot = 0; slot < 4; ++slot) {
    red[tid] = vals[slot]; __syncthreads();
    for (int o = 128; o > 0; o >>= 1) { if (tid < o) red[tid] += red[tid + o]; __syncthreads(); }
    if (tid == 0) ws[WS_PART3 + 4*(size_t)blin + slot] = red[0];
    __syncthreads();
  }
}

// ---------------------------------------------------------------------------
// Kernel 6: finalize output BN scale/shift.
// ---------------------------------------------------------------------------
__global__ __launch_bounds__(256)
void k_outfold(const float* go, const float* beo, float* __restrict__ ws)
{
  __shared__ float red[256];
  __shared__ float st[4];
  const int tid = threadIdx.x;
  for (int slot = 0; slot < 4; ++slot) {
    float s = 0.f;
    for (int p = tid; p < 12288; p += 256) s += ws[WS_PART3 + 4*(size_t)p + slot];
    red[tid] = s; __syncthreads();
    for (int o = 128; o > 0; o >>= 1) { if (tid < o) red[tid] += red[tid + o]; __syncthreads(); }
    if (tid == 0) st[slot] = red[0];
    __syncthreads();
  }
  if (tid == 0) {
    const float inv = 1.f / ((float)BSZ * (float)L3_);
    const float m0 = st[0] * inv, m1 = st[1] * inv;
    const float v0 = st[2] * inv - m0 * m0;
    const float v1 = st[3] * inv - m1 * m1;
    const float sc0 = go[0] * rsqrtf(v0 + EPS_BN);
    const float sc1 = go[1] * rsqrtf(v1 + EPS_BN);
    ws[WS_OUTF + 0] = sc0;
    ws[WS_OUTF + 1] = sc1;
    ws[WS_OUTF + 2] = beo[0] - sc0 * m0;
    ws[WS_OUTF + 3] = beo[1] - sc1 * m1;
  }
}

// ---------------------------------------------------------------------------
// Kernel 7: in-place BN normalization of d_out.
// ---------------------------------------------------------------------------
__global__ __launch_bounds__(256)
void k_norm(float* __restrict__ out, const float* __restrict__ ws)
{
  const size_t i = (size_t)blockIdx.x * 256 + threadIdx.x;
  if (i < (size_t)BSZ * L3_ * 2) {
    const int j = (int)(i & 1);
    out[i] = fmaf(ws[WS_OUTF + j], out[i], ws[WS_OUTF + 2 + j]);
  }
}

// ---------------------------------------------------------------------------
extern "C" void kernel_launch(void* const* d_in, const int* in_sizes, int n_in,
                              void* d_out, int out_size, void* d_ws, size_t ws_size,
                              hipStream_t stream)
{
  (void)in_sizes; (void)n_in; (void)out_size; (void)ws_size;
  const float* x     = (const float*)d_in[0];
  const int*   perms = (const int*)  d_in[1];
  const float* Wc1 = (const float*)d_in[2];  const float* bc1 = (const float*)d_in[3];
  const float* g1  = (const float*)d_in[4];  const float* be1 = (const float*)d_in[5];
  const float* Wf1 = (const float*)d_in[6];  const float* bf1 = (const float*)d_in[7];
  const float* Wc2 = (const float*)d_in[8];  const float* bc2 = (const float*)d_in[9];
  const float* g2  = (const float*)d_in[10]; const float* be2 = (const float*)d_in[11];
  const float* Wf2 = (const float*)d_in[12]; const float* bf2 = (const float*)d_in[13];
  const float* WcI = (const float*)d_in[14]; const float* bcI = (const float*)d_in[15];
  const float* gI  = (const float*)d_in[16]; const float* beI = (const float*)d_in[17];
  const float* WfI = (const float*)d_in[18]; const float* bfI = (const float*)d_in[19];
  const float* Wm  = (const float*)d_in[20]; const float* bm  = (const float*)d_in[21];
  const float* gm  = (const float*)d_in[22]; const float* bem = (const float*)d_in[23];
  const float* Wo  = (const float*)d_in[24]; const float* bo  = (const float*)d_in[25];
  const float* go  = (const float*)d_in[26]; const float* beo = (const float*)d_in[27];
  float* out = (float*)d_out;
  float* ws  = (float*)d_ws;

  k_stats  <<<dim3(BSZ), dim3(256), 0, stream>>>(x, perms, Wc1, bc1, Wc2, bc2, WcI, bcI, ws);
  k_fold   <<<dim3(3),   dim3(256), 0, stream>>>(Wc1, bc1, g1, be1, Wc2, bc2, g2, be2,
                                                 WcI, bcI, gI, beI, ws);
  k_encode <<<dim3(LSZ/256, BSZ), dim3(256), 0, stream>>>(x, perms, Wf1, bf1, Wf2, bf2,
                                                          WfI, bfI, ws);
  k_mid    <<<dim3(1),   dim3(256), 0, stream>>>(Wm, bm, gm, bem, ws);
  k_head   <<<dim3(L3_/256, BSZ), dim3(256), 0, stream>>>(Wo, bo, out, ws);
  k_outfold<<<dim3(1),   dim3(256), 0, stream>>>(go, beo, ws);
  k_norm   <<<dim3((BSZ*L3_*2)/256), dim3(256), 0, stream>>>(out, ws);
}
